// OlmoAttention_28810640621804
// MI455X (gfx1250) — compile-verified, matrix operand order checked by reading, not run on hardware
//
#include <hip/hip_runtime.h>
#include <hip/hip_fp16.h>

// ---------------------------------------------------------------------------
// OlmoAttention for MI455X (gfx1250): LN -> QKV GEMM -> RoPE -> flash attn
// -> out GEMM.  All matmuls via v_wmma_f32_16x16x32_f16 (wave32 WMMA).
// GEMM B-panels staged through LDS with gfx1250 async global->LDS copies
// (double-buffered, ASYNCcnt-tracked, 2x-unrolled so each phase has a
// compile-time LDS buffer).
// ---------------------------------------------------------------------------

typedef __attribute__((ext_vector_type(16))) _Float16 v16h;
typedef __attribute__((ext_vector_type(8)))  _Float16 v8h;
typedef __attribute__((ext_vector_type(8)))  float    v8f;
typedef __attribute__((ext_vector_type(4)))  int      v4i;

#define B_   2
#define S_   2048
#define D_   2048
#define H_   16
#define DH_  128
#define LN_EPS 1e-5f

#define AS_GLOBAL __attribute__((address_space(1)))
#define AS_LDS    __attribute__((address_space(3)))

// Build a 16-element f16 A/B fragment from two contiguous 16-byte chunks.
// Per ISA layout: halves 0-7 are K = g*8 + 0..7, halves 8-15 are K = 16 + g*8 + 0..7.
__device__ __forceinline__ v16h load_frag(const _Float16* p) {
  v8h lo = *(const v8h*)(p);
  v8h hi = *(const v8h*)(p + 16);
  return __builtin_shufflevector(lo, hi, 0,1,2,3,4,5,6,7,8,9,10,11,12,13,14,15);
}

__device__ __forceinline__ v8f wmma16(v16h a, v16h b, v8f c) {
  return __builtin_amdgcn_wmma_f32_16x16x32_f16(
      /*neg_a=*/false, a, /*neg_b=*/false, b,
      /*c_mod=*/(short)0, c, /*reuse_a=*/false, /*reuse_b=*/false);
}

// gfx1250 async global -> LDS copy of 16 bytes (tracked by ASYNCcnt).
__device__ __forceinline__ void async_copy_b128(_Float16* lds, const _Float16* g) {
  __builtin_amdgcn_global_load_async_to_lds_b128(
      (AS_GLOBAL v4i*)g, (AS_LDS v4i*)lds, /*offset=*/0, /*cpol=*/0);
}

__device__ __forceinline__ float redmax16(float t) {
  t = fmaxf(t, __shfl_xor(t, 1, 32));
  t = fmaxf(t, __shfl_xor(t, 2, 32));
  t = fmaxf(t, __shfl_xor(t, 4, 32));
  t = fmaxf(t, __shfl_xor(t, 8, 32));
  return t;
}
__device__ __forceinline__ float redsum16(float t) {
  t += __shfl_xor(t, 1, 32);
  t += __shfl_xor(t, 2, 32);
  t += __shfl_xor(t, 4, 32);
  t += __shfl_xor(t, 8, 32);
  return t;
}

// ---------------------------------------------------------------------------
// 1) LayerNorm (fp32) -> f16 activations.  One row (2048) per 256-thread block.
// ---------------------------------------------------------------------------
__global__ __launch_bounds__(256) void ln_kernel(const float* __restrict__ x,
                                                 _Float16* __restrict__ xo) {
  __shared__ float s_sum[256];
  __shared__ float s_sq[256];
  const int tid = threadIdx.x;
  const float* row = x + (size_t)blockIdx.x * D_;
  float sum = 0.f, sq = 0.f;
#pragma unroll
  for (int i = 0; i < D_ / 256; ++i) {
    float v = row[tid + i * 256];
    sum += v; sq += v * v;
  }
  s_sum[tid] = sum; s_sq[tid] = sq;
  __syncthreads();
  for (int off = 128; off > 0; off >>= 1) {
    if (tid < off) { s_sum[tid] += s_sum[tid + off]; s_sq[tid] += s_sq[tid + off]; }
    __syncthreads();
  }
  const float mean = s_sum[0] * (1.f / D_);
  const float var  = s_sq[0] * (1.f / D_) - mean * mean;
  const float rstd = rsqrtf(var + LN_EPS);
  _Float16* orow = xo + (size_t)blockIdx.x * D_;
#pragma unroll
  for (int i = 0; i < D_ / 256; ++i) {
    float v = row[tid + i * 256];
    orow[tid + i * 256] = (_Float16)((v - mean) * rstd);
  }
}

// ---------------------------------------------------------------------------
// 2) Convert fp32 weight [K,N] to transposed f16 [N,K] so B-fragments are
//    K-contiguous b128 loads.
// ---------------------------------------------------------------------------
__global__ __launch_bounds__(256) void wt_transpose_kernel(const float* __restrict__ w,
                                                           _Float16* __restrict__ wt,
                                                           int K, int N) {
  size_t idx = (size_t)blockIdx.x * blockDim.x + threadIdx.x;
  if (idx >= (size_t)K * N) return;
  int k = (int)(idx / N);
  int n = (int)(idx % N);
  wt[(size_t)n * K + k] = (_Float16)w[idx];
}

// ---------------------------------------------------------------------------
// 3) WMMA GEMM: C[M,N](f32) = A[M,K](f16,row) * Bt[N,K](f16,row == B^T).
//    Wave: 16x64 tile.  Block: 8 waves => 128x64.  The 64x32 B panel is
//    shared by all 8 waves: staged once per block into LDS via async
//    global->LDS b128 copies, double-buffered; K loop unrolled 2x so each
//    phase addresses a compile-time LDS buffer.
//    Requires M%128==0, N%64==0, K%64==0.
// ---------------------------------------------------------------------------
#define BPAD 48   // padded row stride (halves): 96B, 16B-aligned, bank-spread

__global__ __launch_bounds__(256) void wmma_gemm_kernel(const _Float16* __restrict__ A,
                                                        const _Float16* __restrict__ Bt,
                                                        float* __restrict__ C,
                                                        int M, int N, int K) {
  __shared__ alignas(16) _Float16 bsh[2][64][BPAD];

  const int tid  = threadIdx.x;
  const int lane = tid & 31;
  const int wave = tid >> 5;
  const int m0   = blockIdx.y * 128 + wave * 16;
  const int n0   = blockIdx.x * 64;
  const int mrow = lane & 15;   // M (A) or N (B) index within tile
  const int g    = lane >> 4;   // lane group: K offset +8

  // Staging assignment: thread t copies 16B: row sn of the panel, chunk sc.
  const int sn = tid >> 2;      // 0..63  (panel row = B column n0+sn)
  const int sc = tid & 3;       // 0..3   (16B chunk within 32 halves)
  const _Float16* gsrc = Bt + (size_t)(n0 + sn) * K + sc * 8;
  _Float16* lds0 = &bsh[0][sn][sc * 8];
  _Float16* lds1 = &bsh[1][sn][sc * 8];

  // Prologue: stage first B panel (kk = 0) into buffer 0.
  async_copy_b128(lds0, gsrc);

  v8f acc0 = {}, acc1 = {}, acc2 = {}, acc3 = {};
  const _Float16* ap = A + (size_t)(m0 + mrow) * K + g * 8;
  const _Float16* b0p = &bsh[0][mrow][g * 8];
  const _Float16* b1p = &bsh[1][mrow][g * 8];

  for (int kk = 0; kk < K; kk += 64) {
    // A fragments for both phases: wave-private, issue before the barrier so
    // their global latency overlaps the block handoff.
    __builtin_prefetch(ap + kk + 1024, 0, 1);
    v16h af0 = load_frag(ap + kk);
    v16h af1 = load_frag(ap + kk + 32);

    // ---- phase 0: consume buffer 0 (panel kk), prefetch panel kk+32 -> buf 1
    asm volatile("s_wait_asynccnt 0x0" ::: "memory");
    __syncthreads();
    async_copy_b128(lds1, gsrc + kk + 32);
    {
      v16h b0 = load_frag(b0p + ( 0) * BPAD);
      v16h b1 = load_frag(b0p + (16) * BPAD);
      v16h b2 = load_frag(b0p + (32) * BPAD);
      v16h b3 = load_frag(b0p + (48) * BPAD);
      acc0 = wmma16(af0, b0, acc0);
      acc1 = wmma16(af0, b1, acc1);
      acc2 = wmma16(af0, b2, acc2);
      acc3 = wmma16(af0, b3, acc3);
    }

    // ---- phase 1: consume buffer 1 (panel kk+32), prefetch panel kk+64 -> buf 0
    asm volatile("s_wait_asynccnt 0x0" ::: "memory");
    __syncthreads();
    if (kk + 64 < K)
      async_copy_b128(lds0, gsrc + kk + 64);
    {
      v16h b0 = load_frag(b1p + ( 0) * BPAD);
      v16h b1 = load_frag(b1p + (16) * BPAD);
      v16h b2 = load_frag(b1p + (32) * BPAD);
      v16h b3 = load_frag(b1p + (48) * BPAD);
      acc0 = wmma16(af1, b0, acc0);
      acc1 = wmma16(af1, b1, acc1);
      acc2 = wmma16(af1, b2, acc2);
      acc3 = wmma16(af1, b3, acc3);
    }
  }

  // C fragment: VGPR r -> row m0 + r + 8*g, lane%16 -> column.
  float* c0 = C + (size_t)(m0 + 8 * g) * N + n0 + mrow;
#pragma unroll
  for (int r = 0; r < 8; ++r) {
    float* cr = c0 + (size_t)r * N;
    cr[ 0] = acc0[r];
    cr[16] = acc1[r];
    cr[32] = acc2[r];
    cr[48] = acc3[r];
  }
}

// ---------------------------------------------------------------------------
// 4) RoPE + relayout.  qkv f32 [B*S, 3*D] ->
//      qh [B,H,S,DH] f16, kh [B,H,S,DH] f16, vt [B,H,DH,S] f16 (V transposed).
//    One (b,s,h) per 128-thread block.
// ---------------------------------------------------------------------------
__global__ __launch_bounds__(128) void rope_kernel(const float* __restrict__ qkv,
                                                   const int* __restrict__ positions,
                                                   _Float16* __restrict__ qh,
                                                   _Float16* __restrict__ kh,
                                                   _Float16* __restrict__ vt) {
  const int tid = threadIdx.x;
  const int bsh = blockIdx.x;
  const int h   = bsh % H_;
  const int bs  = bsh / H_;
  const int s   = bs % S_;
  const int b   = bs / S_;
  const float pos = (float)positions[bs];

  const float* base  = qkv + (size_t)bs * (3 * D_);
  const float* qbase = base + h * DH_;
  const float* kbase = base + D_ + h * DH_;
  const float* vbase = base + 2 * D_ + h * DH_;

  _Float16* qo = qh + (((size_t)(b * H_ + h)) * S_ + s) * DH_;
  _Float16* ko = kh + (((size_t)(b * H_ + h)) * S_ + s) * DH_;

  if (tid < 64) {
    const int d = tid;
    const float inv_freq = __powf(10000.0f, -(float)d / 64.0f);
    float sn, cs;
    __sincosf(pos * inv_freq, &sn, &cs);
    const float q1 = qbase[d], q2 = qbase[d + 64];
    const float k1 = kbase[d], k2 = kbase[d + 64];
    qo[d]      = (_Float16)(q1 * cs - q2 * sn);
    qo[d + 64] = (_Float16)(q2 * cs + q1 * sn);
    ko[d]      = (_Float16)(k1 * cs - k2 * sn);
    ko[d + 64] = (_Float16)(k2 * cs + k1 * sn);
  }
  // V: transpose into [B,H,DH,S]
  vt[(((size_t)(b * H_ + h)) * DH_ + tid) * S_ + s] = (_Float16)vbase[tid];
}

// ---------------------------------------------------------------------------
// 5) Flash attention.  grid.x = B*H, grid.y = S/128.  Block 256 = 8 waves,
//    each wave owns 16 query rows, streams keys in 32-key steps with online
//    softmax.  Output f16 into attn [B*S, D] (row = b*S+s, col = h*DH+d).
// ---------------------------------------------------------------------------
__global__ __launch_bounds__(256) void flash_kernel(const _Float16* __restrict__ q,
                                                    const _Float16* __restrict__ k,
                                                    const _Float16* __restrict__ vt,
                                                    _Float16* __restrict__ attn) {
  __shared__ alignas(16) _Float16 pshm[8][16][32];
  const int lane = threadIdx.x & 31;
  const int wave = threadIdx.x >> 5;
  const int bh   = blockIdx.x;
  const int q0   = blockIdx.y * 128 + wave * 16;
  const int mrow = lane & 15;
  const int g    = lane >> 4;
  const float scale = 0.08838834764831845f;  // 1/sqrt(128)

  const _Float16* qp = q  + (size_t)bh * S_ * DH_;
  const _Float16* kp = k  + (size_t)bh * S_ * DH_;
  const _Float16* vp = vt + (size_t)bh * DH_ * S_;

  // Preload 4 A-fragments of Q (DH=128 = 4 chunks of 32).
  v16h qf[4];
#pragma unroll
  for (int c = 0; c < 4; ++c)
    qf[c] = load_frag(qp + (size_t)(q0 + mrow) * DH_ + c * 32 + g * 8);

  float m_i[8], l_i[8];
  v8f acc[8];
#pragma unroll
  for (int r = 0; r < 8; ++r) { m_i[r] = -3.0e38f; l_i[r] = 0.f; }
#pragma unroll
  for (int t = 0; t < 8; ++t) acc[t] = (v8f){};

  for (int ks = 0; ks < q0 + 16; ks += 32) {
    // --- scores: Q(16x128) x K^T(128x32) as two 16x16 C-fragments
    v8f s0 = {}, s1 = {};
#pragma unroll
    for (int c = 0; c < 4; ++c) {
      const _Float16* kb = kp + (size_t)(ks + mrow) * DH_ + c * 32 + g * 8;
      s0 = wmma16(qf[c], load_frag(kb), s0);
      s1 = wmma16(qf[c], load_frag(kb + 16 * DH_), s1);
    }

    // --- online softmax update (rows r+8g live in lane group g)
    float alpha[8];
#pragma unroll
    for (int r = 0; r < 8; ++r) {
      const int qrow = q0 + r + 8 * g;
      float x0 = s0[r] * scale;
      float x1 = s1[r] * scale;
      if (ks + mrow      > qrow) x0 = -3.0e38f;   // causal mask
      if (ks + 16 + mrow > qrow) x1 = -3.0e38f;
      float rmax = redmax16(fmaxf(x0, x1));
      float mnew = fmaxf(m_i[r], rmax);
      float a    = __expf(m_i[r] - mnew);
      float p0   = __expf(x0 - mnew);
      float p1   = __expf(x1 - mnew);
      l_i[r] = l_i[r] * a + redsum16(p0 + p1);
      m_i[r] = mnew;
      alpha[r] = a;
      pshm[wave][r + 8 * g][mrow]      = (_Float16)p0;
      pshm[wave][r + 8 * g][16 + mrow] = (_Float16)p1;
    }
#pragma unroll
    for (int t = 0; t < 8; ++t)
#pragma unroll
      for (int r = 0; r < 8; ++r) acc[t][r] *= alpha[r];

    // make the per-wave LDS probs visible before the fragment re-load
    asm volatile("s_wait_dscnt 0x0" ::: "memory");

    // --- P(16x32) x V(32x128): reload P as an A-fragment from LDS
    v16h pf = load_frag(&pshm[wave][mrow][g * 8]);
#pragma unroll
    for (int t = 0; t < 8; ++t) {
      const _Float16* vb = vp + (size_t)(t * 16 + mrow) * S_ + ks + g * 8;
      acc[t] = wmma16(pf, load_frag(vb), acc[t]);
    }
  }

  // --- epilogue: normalize and store f16
  const int b = bh / H_;
  const int h = bh % H_;
  float inv[8];
#pragma unroll
  for (int r = 0; r < 8; ++r) inv[r] = 1.0f / l_i[r];
  _Float16* ob = attn + (size_t)(b * S_ + q0 + 8 * g) * D_ + h * DH_ + mrow;
#pragma unroll
  for (int r = 0; r < 8; ++r) {
    _Float16* orow = ob + (size_t)r * D_;
#pragma unroll
    for (int t = 0; t < 8; ++t)
      orow[t * 16] = (_Float16)(acc[t][r] * inv[r]);
  }
}

// ---------------------------------------------------------------------------
// Launcher.  Inputs: [0]=positions(int), [1]=hidden(f32), [2]=w_qkv(f32),
// [3]=w_out(f32).  Output: f32 [B,S,D].
// ---------------------------------------------------------------------------
extern "C" void kernel_launch(void* const* d_in, const int* in_sizes, int n_in,
                              void* d_out, int out_size, void* d_ws, size_t ws_size,
                              hipStream_t stream) {
  (void)in_sizes; (void)n_in; (void)out_size; (void)ws_size;
  const int* positions = (const int*)d_in[0];
  const float* hidden  = (const float*)d_in[1];
  const float* w_qkv   = (const float*)d_in[2];
  const float* w_out   = (const float*)d_in[3];
  float* out           = (float*)d_out;

  char* ws = (char*)d_ws;
  size_t off = 0;
  _Float16* x_f16  = (_Float16*)(ws + off); off += (size_t)B_ * S_ * D_ * 2;          // 16 MiB
  _Float16* wqkvT  = (_Float16*)(ws + off); off += (size_t)3 * D_ * D_ * 2;           // 24 MiB
  _Float16* woutT  = (_Float16*)(ws + off); off += (size_t)D_ * D_ * 2;               //  8 MiB
  float*    qkv    = (float*)   (ws + off); off += (size_t)B_ * S_ * 3 * D_ * 4;      // 96 MiB
  _Float16* qh     = (_Float16*)(ws + off); off += (size_t)B_ * H_ * S_ * DH_ * 2;    // 16 MiB
  _Float16* kh     = (_Float16*)(ws + off); off += (size_t)B_ * H_ * S_ * DH_ * 2;    // 16 MiB
  _Float16* vt     = (_Float16*)(ws + off); off += (size_t)B_ * H_ * DH_ * S_ * 2;    // 16 MiB
  _Float16* attn   = (_Float16*)(ws + off); off += (size_t)B_ * S_ * D_ * 2;          // 16 MiB

  // 1) LayerNorm -> f16
  ln_kernel<<<B_ * S_, 256, 0, stream>>>(hidden, x_f16);

  // 2) Transpose + cast weights
  {
    size_t n1 = (size_t)D_ * (3 * D_);
    wt_transpose_kernel<<<(unsigned)((n1 + 255) / 256), 256, 0, stream>>>(w_qkv, wqkvT, D_, 3 * D_);
    size_t n2 = (size_t)D_ * D_;
    wt_transpose_kernel<<<(unsigned)((n2 + 255) / 256), 256, 0, stream>>>(w_out, woutT, D_, D_);
  }

  // 3) QKV GEMM: [4096,2048] x [2048,6144] -> f32
  wmma_gemm_kernel<<<dim3((3 * D_) / 64, (B_ * S_) / 128), 256, 0, stream>>>(
      x_f16, wqkvT, qkv, B_ * S_, 3 * D_, D_);

  // 4) RoPE + relayout
  rope_kernel<<<B_ * S_ * H_, 128, 0, stream>>>(qkv, positions, qh, kh, vt);

  // 5) Flash attention
  flash_kernel<<<dim3(B_ * H_, S_ / 128), 256, 0, stream>>>(qh, kh, vt, attn);

  // 6) Output projection: [4096,2048] x [2048,2048] -> f32 out
  wmma_gemm_kernel<<<dim3(D_ / 64, (B_ * S_) / 128), 256, 0, stream>>>(
      attn, woutT, out, B_ * S_, D_, D_);
}
